// S4Model_24481313587915
// MI455X (gfx1250) — matrix operand order repaired and checked by reading, not compiled, not run on hardware
//
#include <hip/hip_runtime.h>
#include <hip/hip_bf16.h>
#include <math.h>
#include <cstdint>

// ---------------- model dims ----------------
#define B_SZ     32
#define SEQ_L    1024
#define D_IN     128
#define D_MODEL  256
#define N_LAYERS 4
#define N_MODES  32      // D_STATE/2 complex conjugate-pair modes
#define D_OUT    10

// GEMM macro-tile
#define BM 128
#define BN 64
#define BK 32

typedef __attribute__((ext_vector_type(16))) _Float16 v16h;
typedef __attribute__((ext_vector_type(8)))  _Float16 v8h;
typedef __attribute__((ext_vector_type(8)))  float    v8f;
typedef __attribute__((ext_vector_type(4)))  unsigned int u32x4;
typedef __attribute__((ext_vector_type(4)))  int      i32x4;
typedef __attribute__((ext_vector_type(8)))  int      i32x8;

// ---------------------------------------------------------------------------
// TDM: issue a 2D tile load (f16 elements) global -> LDS.
// D# packing per CDNA5 ISA 8.3/8.4:
//   group0: [1:0]=count=1, [63:32]=lds_addr, [120:64]=global_addr, [127:126]=type(2)
//   group1: [17:16]=data_size(1 => 2B), [79:48]=tensor_dim0, [111:80]=tensor_dim1,
//           [127:112]=tile_dim0, [143:128]=tile_dim1, [207:160]=tensor_dim0_stride
// ---------------------------------------------------------------------------
__device__ __forceinline__ unsigned lds_off32(const void* p) {
    // LDS aperture: generic addr low 32 bits == LDS byte offset
    return (unsigned)(uintptr_t)p;
}

__device__ __forceinline__ void tdm_load_tile_f16(const void* lds_dst,
                                                  const _Float16* gsrc,
                                                  unsigned tile_k, unsigned tile_rows,
                                                  unsigned tensor_k, unsigned tensor_rows,
                                                  unsigned long long stride_elems) {
    unsigned long long ga = (unsigned long long)(uintptr_t)gsrc;
    u32x4 g0;
    g0.x = 1u;                                              // count=1 (valid), user mode
    g0.y = lds_off32(lds_dst);                              // lds_addr (bytes)
    g0.z = (unsigned)(ga & 0xFFFFFFFFu);                    // global_addr[31:0]
    g0.w = (unsigned)((ga >> 32) & 0x01FFFFFFu) | 0x80000000u; // addr[56:32] | type=2

    i32x8 g1;
    g1[0] = (int)(1u << 16);                                // wg_mask=0, data_size=1 (2B)
    g1[1] = (int)((tensor_k & 0xFFFFu) << 16);              // tensor_dim0[15:0]
    g1[2] = (int)(((tensor_k >> 16) & 0xFFFFu) |
                  ((tensor_rows & 0xFFFFu) << 16));         // dim0[31:16] | dim1[15:0]
    g1[3] = (int)(((tensor_rows >> 16) & 0xFFFFu) |
                  ((tile_k & 0xFFFFu) << 16));              // dim1[31:16] | tile_dim0
    g1[4] = (int)(tile_rows & 0xFFFFu);                     // tile_dim1 | tile_dim2=0
    g1[5] = (int)(stride_elems & 0xFFFFFFFFu);              // dim0_stride[31:0]
    g1[6] = (int)((stride_elems >> 32) & 0xFFFFu);          // dim0_stride[47:32]
    g1[7] = 0;

    i32x4 z4 = {0, 0, 0, 0};
#if defined(__clang_major__) && (__clang_major__ >= 23)
    i32x8 z8 = {0, 0, 0, 0, 0, 0, 0, 0};
    __builtin_amdgcn_tensor_load_to_lds(g0, g1, z4, z4, z8, 0);
#else
    __builtin_amdgcn_tensor_load_to_lds(g0, g1, z4, z4, 0);
#endif
}

// ---------------------------------------------------------------------------
// 1) Precompute per-(layer,h,n): dA = exp(dt*A), w = 2*C*B*(dA-1)/A (ZOH)
// ---------------------------------------------------------------------------
__global__ void s4_precompute(const float* __restrict__ log_dt,
                              const float* __restrict__ A_re_log,
                              const float* __restrict__ A_im,
                              const float* __restrict__ B_re,
                              const float* __restrict__ B_im,
                              const float* __restrict__ C_re,
                              const float* __restrict__ C_im,
                              float* __restrict__ dAr, float* __restrict__ dAi,
                              float* __restrict__ wr,  float* __restrict__ wi) {
    int idx = blockIdx.x * blockDim.x + threadIdx.x;
    if (idx >= N_LAYERS * D_MODEL * N_MODES) return;
    int lh = idx / N_MODES;
    float dt  = __expf(log_dt[lh]);
    float Are = -__expf(A_re_log[idx]);
    float Aim = A_im[idx];
    float e   = __expf(dt * Are);
    float dar = e * __cosf(dt * Aim);
    float dai = e * __sinf(dt * Aim);
    float nr = dar - 1.0f, ni = dai;
    float inv = 1.0f / (Are * Are + Aim * Aim);
    float qr = (nr * Are + ni * Aim) * inv;
    float qi = (ni * Are - nr * Aim) * inv;
    float Br = B_re[idx], Bi = B_im[idx];
    float dBr = Br * qr - Bi * qi;
    float dBi = Br * qi + Bi * qr;
    float Cr = C_re[idx], Ci = C_im[idx];
    dAr[idx] = dar;
    dAi[idx] = dai;
    wr[idx]  = 2.0f * (Cr * dBr - Ci * dBi);
    wi[idx]  = 2.0f * (Cr * dBi + Ci * dBr);
}

// ---------------------------------------------------------------------------
// helpers: f32 -> f16 convert, f32 -> f16 transpose (weights, one-time)
// ---------------------------------------------------------------------------
__global__ void cvt_f32_to_f16(const float* __restrict__ in,
                               _Float16* __restrict__ out, int n) {
    int idx = blockIdx.x * blockDim.x + threadIdx.x;
    if (idx < n) out[idx] = (_Float16)in[idx];
}

__global__ void transpose_f32_to_f16(const float* __restrict__ in,  // K x N
                                     _Float16* __restrict__ out,    // N x K
                                     int K, int N) {
    int idx = blockIdx.x * blockDim.x + threadIdx.x;
    if (idx >= K * N) return;
    int nn = idx / K, kk = idx % K;
    out[idx] = (_Float16)in[kk * N + nn];
}

// ---------------------------------------------------------------------------
// 2) GEMM: C[M,N] = A[M,K] @ Bt[N,K]^T + bias, f16 WMMA, f32 accumulate.
//    TDM double-buffered LDS staging; block = 8 waves = 128x64 macro-tile;
//    each wave: 16 rows x 64 cols = 4 WMMA tiles sharing one A fragment.
//    Fragment layouts per ISA 7.12.2 (A: K {0..7,16..23}/{8..15,24..31} by
//    lane-half; B: K {0..7,16..23}/{8..15,24..31} by lane-half, one column per lane).
// ---------------------------------------------------------------------------
__global__ void gemm_wmma_tdm(const _Float16* __restrict__ A,   // M x K
                              const _Float16* __restrict__ Bt,  // N x K
                              const float* __restrict__ bias,
                              float* __restrict__ C,
                              int M, int N, int K) {
    __shared__ _Float16 sA[2][BM * BK];   // 2 x 8 KB
    __shared__ _Float16 sB[2][BN * BK];   // 2 x 4 KB

    const int lane = threadIdx.x & 31;
    const int wv   = threadIdx.x >> 5;
    const int row0 = blockIdx.y * BM;
    const int col0 = blockIdx.x * BN;

    const int m   = lane & 15;
    const int hi  = lane >> 4;
    const int kba = hi ? 8 : 0;
    const int kbb = hi ? 8 : 0;
    const int n   = lane & 15;

    v8f acc[4] = {};

    if (wv == 0) {
        tdm_load_tile_f16(&sA[0][0], A + (size_t)row0 * K, BK, BM, K, M, K);
        tdm_load_tile_f16(&sB[0][0], Bt + (size_t)col0 * K, BK, BN, K, N, K);
        __builtin_amdgcn_s_wait_tensorcnt((short)0);
    }
    __syncthreads();

    int p = 0;
    for (int k0 = 0; k0 < K; k0 += BK) {
        const bool has_next = (k0 + BK) < K;
        if (has_next && wv == 0) {
            tdm_load_tile_f16(&sA[p ^ 1][0], A + (size_t)row0 * K + (k0 + BK),
                              BK, BM, K, M, K);
            tdm_load_tile_f16(&sB[p ^ 1][0], Bt + (size_t)col0 * K + (k0 + BK),
                              BK, BN, K, N, K);
        }

        // A fragment (this wave's 16-row subtile), two aligned 16B LDS reads
        const _Float16* abase = &sA[p][(wv * 16 + m) * BK];
        v8h alo = *(const v8h*)(abase + kba);
        v8h ahi = *(const v8h*)(abase + 16 + kba);
        v16h af;
#pragma unroll
        for (int i = 0; i < 8; ++i) { af[i] = alo[i]; af[8 + i] = ahi[i]; }

#pragma unroll
        for (int c = 0; c < 4; ++c) {
            const _Float16* bbase = &sB[p][(c * 16 + n) * BK + kbb];
            v8h blo = *(const v8h*)(bbase);
            v8h bhi = *(const v8h*)(bbase + 16);
            v16h bf;
#pragma unroll
            for (int i = 0; i < 8; ++i) { bf[i] = blo[i]; bf[8 + i] = bhi[i]; }
            acc[c] = __builtin_amdgcn_wmma_f32_16x16x32_f16(
                false, af, false, bf, (short)0, acc[c], false, false);
        }

        __syncthreads();              // all waves done reading buffer p
        if (has_next) {
            if (wv == 0) __builtin_amdgcn_s_wait_tensorcnt((short)0);
            __syncthreads();          // publish buffer p^1
        }
        p ^= 1;
    }

    const int mb = hi ? 8 : 0;
#pragma unroll
    for (int c = 0; c < 4; ++c) {
        const int col = col0 + c * 16 + n;
        const float bv = bias[col];
#pragma unroll
        for (int r = 0; r < 8; ++r)
            C[(size_t)(row0 + wv * 16 + mb + r) * N + col] = acc[c][r] + bv;
    }
}

// ---------------------------------------------------------------------------
// 3) S4D recurrence + D*u skip + GELU(tanh). One wave per (b,h) channel,
//    lane = complex mode. Writes f16 (feeds next GEMM's A operand).
// ---------------------------------------------------------------------------
__global__ void s4_scan(const float* __restrict__ hbuf,   // (B, L, H) f32
                        const float* __restrict__ dAr,
                        const float* __restrict__ dAi,
                        const float* __restrict__ wr,
                        const float* __restrict__ wi,
                        const float* __restrict__ Dp,
                        _Float16* __restrict__ yg) {      // (B, L, H) f16
    const int lane = threadIdx.x & 31;
    const int wid  = blockIdx.x * (blockDim.x >> 5) + (threadIdx.x >> 5);
    if (wid >= B_SZ * D_MODEL) return;
    const int b  = wid >> 8;
    const int hh = wid & 255;

    const int pidx = hh * N_MODES + lane;
    const float ar = dAr[pidx], ai = dAi[pidx];
    const float wre = wr[pidx], wim = wi[pidx];
    const float Dv = Dp[hh];

    const float*    up = hbuf + ((size_t)b * SEQ_L) * D_MODEL + hh;
    _Float16*       yp = yg   + ((size_t)b * SEQ_L) * D_MODEL + hh;

    float tr = 0.0f, ti = 0.0f;
    for (int l0 = 0; l0 < SEQ_L; l0 += 32) {
        float ysave = 0.0f;
#pragma unroll 4
        for (int j = 0; j < 32; ++j) {
            const float u = up[(size_t)(l0 + j) * D_MODEL]; // wave-uniform load
            const float ntr = ar * tr - ai * ti + u;
            const float nti = ar * ti + ai * tr;
            tr = ntr; ti = nti;
            float c = wre * tr - wim * ti;
            c += __shfl_xor(c, 16, 32);
            c += __shfl_xor(c, 8, 32);
            c += __shfl_xor(c, 4, 32);
            c += __shfl_xor(c, 2, 32);
            c += __shfl_xor(c, 1, 32);
            const float y = c + Dv * u;
            const float g = 0.5f * y *
                (1.0f + tanhf(0.7978845608f * (y + 0.044715f * y * y * y)));
            if (lane == j) ysave = g;
        }
        yp[(size_t)(l0 + lane) * D_MODEL] = (_Float16)ysave;
    }
}

// ---------------------------------------------------------------------------
// 4) GLU + residual + LayerNorm fused, block = one token (256 threads).
// ---------------------------------------------------------------------------
__global__ void glu_res_ln(const float* __restrict__ z,   // (B*L, 2H)
                           float* __restrict__ hbuf,      // (B*L, H) in/out
                           const float* __restrict__ lnw,
                           const float* __restrict__ lnb) {
    __shared__ float red[D_MODEL];
    const int t = threadIdx.x;
    const size_t tok = blockIdx.x;
    const float a  = z[tok * (2 * D_MODEL) + t];
    const float gt = z[tok * (2 * D_MODEL) + D_MODEL + t];
    const float g  = a * (1.0f / (1.0f + __expf(-gt))) + hbuf[tok * D_MODEL + t];

    red[t] = g;
    __syncthreads();
#pragma unroll
    for (int s = D_MODEL / 2; s > 0; s >>= 1) {
        if (t < s) red[t] += red[t + s];
        __syncthreads();
    }
    const float mean = red[0] * (1.0f / D_MODEL);
    __syncthreads();
    const float d = g - mean;
    red[t] = d * d;
    __syncthreads();
#pragma unroll
    for (int s = D_MODEL / 2; s > 0; s >>= 1) {
        if (t < s) red[t] += red[t + s];
        __syncthreads();
    }
    const float var = red[0] * (1.0f / D_MODEL);
    hbuf[tok * D_MODEL + t] = d * rsqrtf(var + 1e-5f) * lnw[t] + lnb[t];
}

// ---------------------------------------------------------------------------
// 5) Decoder (N=10: scalar; 0.4% of FLOPs)
// ---------------------------------------------------------------------------
__global__ void decoder(const float* __restrict__ hbuf,
                        const float* __restrict__ dec_w,
                        const float* __restrict__ dec_b,
                        float* __restrict__ out) {
    const int idx = blockIdx.x * blockDim.x + threadIdx.x;
    if (idx >= B_SZ * SEQ_L * D_OUT) return;
    const int o = idx % D_OUT;
    const size_t tok = idx / D_OUT;
    float acc = dec_b[o];
    const float* hp = hbuf + tok * D_MODEL;
#pragma unroll 8
    for (int k = 0; k < D_MODEL; ++k)
        acc += hp[k] * dec_w[k * D_OUT + o];
    out[idx] = acc;
}

// ---------------------------------------------------------------------------
extern "C" void kernel_launch(void* const* d_in, const int* in_sizes, int n_in,
                              void* d_out, int out_size, void* d_ws, size_t ws_size,
                              hipStream_t stream) {
    const float* x        = (const float*)d_in[0];
    const float* enc_w    = (const float*)d_in[1];
    const float* enc_b    = (const float*)d_in[2];
    const float* log_dt   = (const float*)d_in[3];
    const float* A_re_log = (const float*)d_in[4];
    const float* A_im     = (const float*)d_in[5];
    const float* B_re     = (const float*)d_in[6];
    const float* B_im     = (const float*)d_in[7];
    const float* C_re     = (const float*)d_in[8];
    const float* C_im     = (const float*)d_in[9];
    const float* Dp       = (const float*)d_in[10];
    const float* ln_w     = (const float*)d_in[11];
    const float* ln_b     = (const float*)d_in[12];
    const float* out_w    = (const float*)d_in[13];
    const float* out_b    = (const float*)d_in[14];
    const float* dec_w    = (const float*)d_in[15];
    const float* dec_b    = (const float*)d_in[16];
    float* out = (float*)d_out;

    // ---- workspace layout (byte offsets, all 16B aligned) ----
    uint8_t* w8 = (uint8_t*)d_ws;
    float*     hb  = (float*)(w8);                            //  0..32MB  (B*L,H) f32
    float*     zb  = (float*)(w8 + (32ull << 20));            // 32..96MB  (B*L,2H) f32
    _Float16*  xh  = (_Float16*)(w8 + (96ull << 20));         // 96..104MB x in f16
    _Float16*  ygh = (_Float16*)(w8 + (104ull << 20));        // 104..120MB gelu(y) f16
    _Float16*  ewT = (_Float16*)(w8 + (120ull << 20));        // enc_w^T f16 (256x128)
    _Float16*  owT = ewT + (size_t)D_MODEL * D_IN;            // out_w^T f16 (4x512x256)
    float*     par = (float*)(w8 + (122ull << 20));           // S4D params
    const size_t PSZ = (size_t)N_LAYERS * D_MODEL * N_MODES;  // 32768
    float* dAr = par;
    float* dAi = par + PSZ;
    float* wr  = par + 2 * PSZ;
    float* wi  = par + 3 * PSZ;

    const int M = B_SZ * SEQ_L;   // 32768 tokens

    // 0) one-time operand prep: f16 activations, transposed f16 weights
    cvt_f32_to_f16<<<(M * D_IN + 255) / 256, 256, 0, stream>>>(x, xh, M * D_IN);
    transpose_f32_to_f16<<<(D_IN * D_MODEL + 255) / 256, 256, 0, stream>>>(
        enc_w, ewT, D_IN, D_MODEL);
    for (int i = 0; i < N_LAYERS; ++i)
        transpose_f32_to_f16<<<(D_MODEL * 2 * D_MODEL + 255) / 256, 256, 0, stream>>>(
            out_w + (size_t)i * D_MODEL * 2 * D_MODEL,
            owT + (size_t)i * 2 * D_MODEL * D_MODEL, D_MODEL, 2 * D_MODEL);

    // 1) discretize S4D params
    s4_precompute<<<(int)((PSZ + 255) / 256), 256, 0, stream>>>(
        log_dt, A_re_log, A_im, B_re, B_im, C_re, C_im, dAr, dAi, wr, wi);

    // 2) encoder GEMM: h = x @ enc_w + enc_b   (32768x128 @ 128x256)
    gemm_wmma_tdm<<<dim3(D_MODEL / BN, M / BM), 256, 0, stream>>>(
        xh, ewT, enc_b, hb, M, D_MODEL, D_IN);

    // 3) layer stack
    for (int i = 0; i < N_LAYERS; ++i) {
        const size_t po = (size_t)i * D_MODEL * N_MODES;
        s4_scan<<<(B_SZ * D_MODEL) / 8, 256, 0, stream>>>(
            hb, dAr + po, dAi + po, wr + po, wi + po,
            Dp + (size_t)i * D_MODEL, ygh);
        gemm_wmma_tdm<<<dim3((2 * D_MODEL) / BN, M / BM), 256, 0, stream>>>(
            ygh, owT + (size_t)i * 2 * D_MODEL * D_MODEL,
            out_b + (size_t)i * 2 * D_MODEL, zb, M, 2 * D_MODEL, D_MODEL);
        glu_res_ln<<<M, D_MODEL, 0, stream>>>(
            zb, hb, ln_w + (size_t)i * D_MODEL, ln_b + (size_t)i * D_MODEL);
    }

    // 4) decoder
    decoder<<<(B_SZ * SEQ_L * D_OUT + 255) / 256, 256, 0, stream>>>(
        hb, dec_w, dec_b, out);
}
